// VitBlock_33036888441258
// MI455X (gfx1250) — matrix operand order repaired and checked by reading, not compiled, not run on hardware
//
#include <hip/hip_runtime.h>
#include <math.h>

typedef __bf16 bf16_t;
typedef __attribute__((ext_vector_type(16))) __bf16 v16bf;
typedef __attribute__((ext_vector_type(8)))  __bf16 v8bf;
typedef __attribute__((ext_vector_type(8)))  float  v8f;

#define WMMA_BF16(A, B, C) \
  __builtin_amdgcn_wmma_f32_16x16x32_bf16(false, (A), false, (B), (short)0, (C), false, false)

static constexpr int Bsz = 16, S = 1024, D = 768, H = 12, DH = 64, MLPD = 3072;
static constexpr int ROWS = Bsz * S;  // 16384

// ---- gfx1250 async global->LDS copy (ASYNCcnt-tracked, no VGPR staging) ----
__device__ inline void async_b128(const bf16_t* g, bf16_t* l) {
  asm volatile("global_load_async_to_lds_b128 %0, %1, off"
               :: "v"((unsigned)(unsigned long long)l),
                  "v"((unsigned long long)g)
               : "memory");
}
__device__ inline void wait_async0() {
  asm volatile("s_wait_asynccnt 0x0" ::: "memory");
}

__device__ inline bf16_t f2bf(float f) {
  unsigned u = __float_as_uint(f);
  unsigned r = (u + 0x7fffu + ((u >> 16) & 1u)) >> 16;
  unsigned short us = (unsigned short)r;
  bf16_t b;
  __builtin_memcpy(&b, &us, 2);
  return b;
}

__device__ inline v8f vzero8() {
  v8f z;
#pragma unroll
  for (int i = 0; i < 8; ++i) z[i] = 0.0f;
  return z;
}

// ---- WMMA fragment loaders (wave32, 16-bit operands) ----
// A 16x32 (MxK): lane holds row M=l%16; half=l/16 -> K {half*8..+7} and {16+half*8..+7}
__device__ inline v16bf frag_a(const bf16_t* base, int ld, int ko) {
  const int lane = threadIdx.x & 31;
  const int m = lane & 15, half = lane >> 4;
  const bf16_t* p = base + m * ld + ko;
  v8bf lo = *(const v8bf*)(p + half * 8);
  v8bf hi = *(const v8bf*)(p + 16 + half * 8);
  v16bf r;
#pragma unroll
  for (int i = 0; i < 8; ++i) { r[i] = lo[i]; r[8 + i] = hi[i]; }
  return r;
}

// B 32x16 (KxN) given as Bt[n][k] row-major: lane holds col N=l%16; K {h2*8..+7} and {16+h2*8..+7}
__device__ inline v16bf frag_b(const bf16_t* base, int ld, int ko) {
  const int lane = threadIdx.x & 31;
  const int n = lane & 15, h2 = lane >> 4;
  const bf16_t* p = base + n * ld + ko;
  v8bf lo = *(const v8bf*)(p + h2 * 8);
  v8bf hi = *(const v8bf*)(p + 16 + h2 * 8);
  v16bf r;
#pragma unroll
  for (int i = 0; i < 8; ++i) { r[i] = lo[i]; r[8 + i] = hi[i]; }
  return r;
}

// ---------------- weight fp32 -> bf16 ----------------
__global__ void cvt_f32_bf16(const float* __restrict__ src, bf16_t* __restrict__ dst, int n) {
  int i = blockIdx.x * blockDim.x + threadIdx.x;
  if (i < n) dst[i] = f2bf(src[i]);
}

// ---------------- residual + LayerNorm (one row / block of 256) ----------------
__global__ __launch_bounds__(256) void resid_ln_kernel(
    const float* __restrict__ x, const float* __restrict__ add, float* __restrict__ resid_out,
    const float* __restrict__ g, const float* __restrict__ b, bf16_t* __restrict__ y) {
  const int row = blockIdx.x;
  const int tid = threadIdx.x;
  const float* xr = x + (size_t)row * D;
  float v0 = xr[tid], v1 = xr[tid + 256], v2 = xr[tid + 512];
  if (add) {
    const float* ar = add + (size_t)row * D;
    v0 += ar[tid]; v1 += ar[tid + 256]; v2 += ar[tid + 512];
  }
  if (resid_out) {
    float* rr = resid_out + (size_t)row * D;
    rr[tid] = v0; rr[tid + 256] = v1; rr[tid + 512] = v2;
  }
  float s = v0 + v1 + v2;
  float ss = v0 * v0 + v1 * v1 + v2 * v2;
#pragma unroll
  for (int o = 16; o > 0; o >>= 1) {
    s += __shfl_xor(s, o, 32);
    ss += __shfl_xor(ss, o, 32);
  }
  __shared__ float ps[8], pss[8];
  const int wid = tid >> 5;
  if ((tid & 31) == 0) { ps[wid] = s; pss[wid] = ss; }
  __syncthreads();
  s = 0.f; ss = 0.f;
#pragma unroll
  for (int i = 0; i < 8; ++i) { s += ps[i]; ss += pss[i]; }
  const float mu = s * (1.0f / D);
  const float var = ss * (1.0f / D) - mu * mu;
  const float rs = rsqrtf(var + 1e-5f);
  bf16_t* yr = y + (size_t)row * D;
  yr[tid]       = f2bf((v0 - mu) * rs * g[tid]       + b[tid]);
  yr[tid + 256] = f2bf((v1 - mu) * rs * g[tid + 256] + b[tid + 256]);
  yr[tid + 512] = f2bf((v2 - mu) * rs * g[tid + 512] + b[tid + 512]);
}

// ---------------- QKV projection (per-head 64x64), WMMA ----------------
__global__ __launch_bounds__(128) void qkv_kernel(
    const bf16_t* __restrict__ xln,
    const bf16_t* __restrict__ wq, const bf16_t* __restrict__ wk, const bf16_t* __restrict__ wv,
    const float* __restrict__ bq, const float* __restrict__ bk, const float* __restrict__ bv,
    bf16_t* __restrict__ qo, bf16_t* __restrict__ ko, bf16_t* __restrict__ vo) {
  const int mbase = blockIdx.x * 64;
  const int h = blockIdx.y;
  const int tid = threadIdx.x;
  __shared__ bf16_t Xs[64][72];
  __shared__ bf16_t Ws[3][64][72];

#pragma unroll
  for (int it = 0; it < 4; ++it) {
    int idx = (it * 128 + tid) * 8;
    int r = idx >> 6, c = idx & 63;
    async_b128(xln + (size_t)(mbase + r) * D + h * DH + c, &Xs[r][c]);
    async_b128(wq + (size_t)h * 4096 + idx, &Ws[0][r][c]);
    async_b128(wk + (size_t)h * 4096 + idx, &Ws[1][r][c]);
    async_b128(wv + (size_t)h * 4096 + idx, &Ws[2][r][c]);
  }
  wait_async0();
  __syncthreads();

  const int wid = tid >> 5;
  const int lane = tid & 31;
  const int m16 = lane & 15, half = lane >> 4;
  v16bf a0 = frag_a(&Xs[wid * 16][0], 72, 0);
  v16bf a1 = frag_a(&Xs[wid * 16][0], 72, 32);

#pragma unroll
  for (int mat = 0; mat < 3; ++mat) {
    const float* bias = (mat == 0 ? bq : mat == 1 ? bk : bv) + h * DH;
    bf16_t* outp = (mat == 0 ? qo : mat == 1 ? ko : vo);
    v8f acc[4];
    v16bf bfr[4];
#pragma unroll
    for (int nt = 0; nt < 4; ++nt) bfr[nt] = frag_b(&Ws[mat][nt * 16][0], 72, 0);
#pragma unroll
    for (int nt = 0; nt < 4; ++nt) { acc[nt] = vzero8(); acc[nt] = WMMA_BF16(a0, bfr[nt], acc[nt]); }
#pragma unroll
    for (int nt = 0; nt < 4; ++nt) bfr[nt] = frag_b(&Ws[mat][nt * 16][0], 72, 32);
#pragma unroll
    for (int nt = 0; nt < 4; ++nt) acc[nt] = WMMA_BF16(a1, bfr[nt], acc[nt]);
#pragma unroll
    for (int nt = 0; nt < 4; ++nt) {
      const int col = nt * 16 + m16;
      const float bs = bias[col];
#pragma unroll
      for (int j = 0; j < 8; ++j) {
        int m = mbase + wid * 16 + j + 8 * half;
        int bb = m >> 10, s = m & 1023;
        outp[(((size_t)bb * H + h) * S + s) * DH + col] = f2bf(acc[nt][j] + bs);
      }
    }
  }
}

// ---------------- flash attention: grid(B*H, S/128), 8 waves x 16 rows ----------------
__global__ __launch_bounds__(256) void attn_kernel(
    const bf16_t* __restrict__ q, const bf16_t* __restrict__ k, const bf16_t* __restrict__ v,
    float* __restrict__ attn_out) {
  const int bh = blockIdx.x;
  const int qbase = blockIdx.y * 128;
  const int tid = threadIdx.x;
  const int wid = tid >> 5, lane = tid & 31;
  const int m16 = lane & 15, half = lane >> 4;

  __shared__ bf16_t Ks[32][72];     // [key][dh]
  __shared__ bf16_t Vt[64][40];     // [dh][key]  (transposed)
  __shared__ bf16_t Ps[8][16][40];  // per-wave P staging

  const bf16_t* qrow = q + ((size_t)bh * S + qbase + wid * 16) * DH;
  v16bf qa0 = frag_a(qrow, DH, 0);
  v16bf qa1 = frag_a(qrow, DH, 32);

  float rmax[8], rsum[8];
  v8f o[4];
#pragma unroll
  for (int j = 0; j < 8; ++j) { rmax[j] = -1e30f; rsum[j] = 0.f; }
#pragma unroll
  for (int nt = 0; nt < 4; ++nt) o[nt] = vzero8();

  for (int kt = 0; kt < S / 32; ++kt) {
    __syncthreads();
    {
      int idx = tid * 8;
      int r = idx >> 6, c = idx & 63;
      async_b128(k + ((size_t)bh * S + kt * 32 + r) * DH + c, &Ks[r][c]);
      v8bf vv = *(const v8bf*)(v + ((size_t)bh * S + kt * 32 + r) * DH + c);
#pragma unroll
      for (int i = 0; i < 8; ++i) Vt[c + i][r] = vv[i];
    }
    wait_async0();
    __syncthreads();

    v8f s0 = vzero8(), s1 = vzero8();
    {
      v16bf kb0 = frag_b(&Ks[0][0], 72, 0);
      v16bf kb1 = frag_b(&Ks[0][0], 72, 32);
      v16bf kb2 = frag_b(&Ks[16][0], 72, 0);
      v16bf kb3 = frag_b(&Ks[16][0], 72, 32);
      s0 = WMMA_BF16(qa0, kb0, s0);
      s0 = WMMA_BF16(qa1, kb1, s0);
      s1 = WMMA_BF16(qa0, kb2, s1);
      s1 = WMMA_BF16(qa1, kb3, s1);
    }
#pragma unroll
    for (int j = 0; j < 8; ++j) { s0[j] *= 0.125f; s1[j] *= 0.125f; }

#pragma unroll
    for (int j = 0; j < 8; ++j) {
      float mx = fmaxf(s0[j], s1[j]);
#pragma unroll
      for (int off = 8; off > 0; off >>= 1) mx = fmaxf(mx, __shfl_xor(mx, off, 16));
      const float nm = fmaxf(rmax[j], mx);
      const float p0 = __expf(s0[j] - nm);
      const float p1 = __expf(s1[j] - nm);
      const float corr = __expf(rmax[j] - nm);
      float rs = p0 + p1;
#pragma unroll
      for (int off = 8; off > 0; off >>= 1) rs += __shfl_xor(rs, off, 16);
      rsum[j] = rsum[j] * corr + rs;
      rmax[j] = nm;
#pragma unroll
      for (int nt = 0; nt < 4; ++nt) o[nt][j] *= corr;
      Ps[wid][j + 8 * half][m16] = f2bf(p0);
      Ps[wid][j + 8 * half][16 + m16] = f2bf(p1);
    }
    asm volatile("s_wait_dscnt 0x0" ::: "memory");  // P writes visible to wave's reads

    v16bf pa = frag_a(&Ps[wid][0][0], 40, 0);
    v16bf vb0 = frag_b(&Vt[0][0], 40, 0);
    v16bf vb1 = frag_b(&Vt[16][0], 40, 0);
    v16bf vb2 = frag_b(&Vt[32][0], 40, 0);
    v16bf vb3 = frag_b(&Vt[48][0], 40, 0);
    o[0] = WMMA_BF16(pa, vb0, o[0]);
    o[1] = WMMA_BF16(pa, vb1, o[1]);
    o[2] = WMMA_BF16(pa, vb2, o[2]);
    o[3] = WMMA_BF16(pa, vb3, o[3]);
  }

  const int bb = bh / H, h = bh % H;
#pragma unroll
  for (int j = 0; j < 8; ++j) {
    const float inv = 1.0f / rsum[j];
    const int s = qbase + wid * 16 + j + 8 * half;
#pragma unroll
    for (int nt = 0; nt < 4; ++nt) {
      attn_out[((size_t)bb * S + s) * D + h * DH + nt * 16 + m16] = o[nt][j] * inv;
    }
  }
}

// ---------------- MLP1: [16384,768] x W_in^T -> gelu -> bf16 [16384,3072] ----------------
__global__ __launch_bounds__(256) void mlp1_kernel(
    const bf16_t* __restrict__ x, const bf16_t* __restrict__ w,
    const float* __restrict__ bias, bf16_t* __restrict__ y) {
  const int mb = blockIdx.x * 128, nb = blockIdx.y * 128;
  const int tid = threadIdx.x;
  const int wid = tid >> 5, lane = tid & 31;
  const int m16 = lane & 15, half = lane >> 4;
  __shared__ bf16_t Xs[2][128][72];
  __shared__ bf16_t Wt[2][128][72];

  v8f acc[8];
#pragma unroll
  for (int nt = 0; nt < 8; ++nt) acc[nt] = vzero8();

  auto issue = [&](int kk, int buf) {
    const int k0 = kk * 64;
#pragma unroll
    for (int it = 0; it < 4; ++it) {
      int idx = (it * 256 + tid) * 8;
      int r = idx >> 6, c = idx & 63;
      async_b128(x + (size_t)(mb + r) * D + k0 + c, &Xs[buf][r][c]);
      async_b128(w + (size_t)(nb + r) * D + k0 + c, &Wt[buf][r][c]);
    }
  };

  constexpr int NK = D / 64;  // 12
  issue(0, 0);
  wait_async0();
  __syncthreads();

  for (int kk = 0; kk < NK; ++kk) {
    const int cur = kk & 1;
    if (kk + 1 < NK) issue(kk + 1, cur ^ 1);  // prefetch next chunk (async, other buffer)

    v16bf a0 = frag_a(&Xs[cur][wid * 16][0], 72, 0);
    v16bf a1 = frag_a(&Xs[cur][wid * 16][0], 72, 32);
    v16bf bfr[8];
#pragma unroll
    for (int nt = 0; nt < 8; ++nt) bfr[nt] = frag_b(&Wt[cur][nt * 16][0], 72, 0);
#pragma unroll
    for (int nt = 0; nt < 8; ++nt) acc[nt] = WMMA_BF16(a0, bfr[nt], acc[nt]);
#pragma unroll
    for (int nt = 0; nt < 8; ++nt) bfr[nt] = frag_b(&Wt[cur][nt * 16][0], 72, 32);
#pragma unroll
    for (int nt = 0; nt < 8; ++nt) acc[nt] = WMMA_BF16(a1, bfr[nt], acc[nt]);

    if (kk + 1 < NK) { wait_async0(); __syncthreads(); }
  }

#pragma unroll
  for (int nt = 0; nt < 8; ++nt) {
    const int col = nb + nt * 16 + m16;
    const float bs = bias[col];
#pragma unroll
    for (int j = 0; j < 8; ++j) {
      const int m = mb + wid * 16 + j + 8 * half;
      float t = acc[nt][j] + bs;
      t = 0.5f * t * (1.0f + erff(t * 0.70710678118654752f));  // exact gelu
      y[(size_t)m * MLPD + col] = f2bf(t);
    }
  }
}

// ---------------- MLP2: [16384,3072] x W_out^T + bias + residual -> d_out ----------------
__global__ __launch_bounds__(256) void mlp2_kernel(
    const bf16_t* __restrict__ x, const bf16_t* __restrict__ w,
    const float* __restrict__ bias, float* __restrict__ out) {
  const int mb = blockIdx.x * 128, nb = blockIdx.y * 64;
  const int tid = threadIdx.x;
  const int wid = tid >> 5, lane = tid & 31;
  const int m16 = lane & 15, half = lane >> 4;
  __shared__ bf16_t Xs[2][128][72];
  __shared__ bf16_t Wt[2][64][72];

  v8f acc[4];
#pragma unroll
  for (int nt = 0; nt < 4; ++nt) acc[nt] = vzero8();

  auto issue = [&](int kk, int buf) {
    const int k0 = kk * 64;
#pragma unroll
    for (int it = 0; it < 4; ++it) {
      int idx = (it * 256 + tid) * 8;
      int r = idx >> 6, c = idx & 63;
      async_b128(x + (size_t)(mb + r) * MLPD + k0 + c, &Xs[buf][r][c]);
    }
#pragma unroll
    for (int it = 0; it < 2; ++it) {
      int idx = (it * 256 + tid) * 8;
      int r = idx >> 6, c = idx & 63;
      async_b128(w + (size_t)(nb + r) * MLPD + k0 + c, &Wt[buf][r][c]);
    }
  };

  constexpr int NK = MLPD / 64;  // 48
  issue(0, 0);
  wait_async0();
  __syncthreads();

  for (int kk = 0; kk < NK; ++kk) {
    const int cur = kk & 1;
    if (kk + 1 < NK) issue(kk + 1, cur ^ 1);

    v16bf a0 = frag_a(&Xs[cur][wid * 16][0], 72, 0);
    v16bf a1 = frag_a(&Xs[cur][wid * 16][0], 72, 32);
    v16bf bfr[4];
#pragma unroll
    for (int nt = 0; nt < 4; ++nt) bfr[nt] = frag_b(&Wt[cur][nt * 16][0], 72, 0);
#pragma unroll
    for (int nt = 0; nt < 4; ++nt) acc[nt] = WMMA_BF16(a0, bfr[nt], acc[nt]);
#pragma unroll
    for (int nt = 0; nt < 4; ++nt) bfr[nt] = frag_b(&Wt[cur][nt * 16][0], 72, 32);
#pragma unroll
    for (int nt = 0; nt < 4; ++nt) acc[nt] = WMMA_BF16(a1, bfr[nt], acc[nt]);

    if (kk + 1 < NK) { wait_async0(); __syncthreads(); }
  }

#pragma unroll
  for (int nt = 0; nt < 4; ++nt) {
    const int col = nb + nt * 16 + m16;
    const float bs = bias[col];
#pragma unroll
    for (int j = 0; j < 8; ++j) {
      const int m = mb + wid * 16 + j + 8 * half;
      const size_t idx = (size_t)m * D + col;
      out[idx] = out[idx] + acc[nt][j] + bs;  // residual already in out
    }
  }
}

// ---------------- host launcher ----------------
extern "C" void kernel_launch(void* const* d_in, const int* in_sizes, int n_in,
                              void* d_out, int out_size, void* d_ws, size_t ws_size,
                              hipStream_t stream) {
  const float* patch = (const float*)d_in[0];
  const float* g1 = (const float*)d_in[1];
  const float* b1 = (const float*)d_in[2];
  const float* Wq = (const float*)d_in[3];
  const float* bq = (const float*)d_in[4];
  const float* Wk = (const float*)d_in[5];
  const float* bk = (const float*)d_in[6];
  const float* Wv = (const float*)d_in[7];
  const float* bv = (const float*)d_in[8];
  const float* g2 = (const float*)d_in[9];
  const float* b2 = (const float*)d_in[10];
  const float* W_in = (const float*)d_in[11];
  const float* b_in = (const float*)d_in[12];
  const float* W_out = (const float*)d_in[13];
  const float* b_out = (const float*)d_in[14];
  float* out = (float*)d_out;

  char* ws = (char*)d_ws;
  size_t off = 0;
  auto alloc = [&](size_t bytes) {
    void* p = ws + off;
    off = (off + bytes + 255) & ~(size_t)255;
    return p;
  };
  bf16_t* xln  = (bf16_t*)alloc((size_t)ROWS * D * 2);
  bf16_t* qb   = (bf16_t*)alloc((size_t)ROWS * D * 2);
  bf16_t* kb   = (bf16_t*)alloc((size_t)ROWS * D * 2);
  bf16_t* vb   = (bf16_t*)alloc((size_t)ROWS * D * 2);
  float*  attn = (float*) alloc((size_t)ROWS * D * 4);
  bf16_t* h2   = (bf16_t*)alloc((size_t)ROWS * D * 2);
  bf16_t* hmid = (bf16_t*)alloc((size_t)ROWS * MLPD * 2);
  bf16_t* wqb  = (bf16_t*)alloc((size_t)H * DH * DH * 2);
  bf16_t* wkb  = (bf16_t*)alloc((size_t)H * DH * DH * 2);
  bf16_t* wvb  = (bf16_t*)alloc((size_t)H * DH * DH * 2);
  bf16_t* winb = (bf16_t*)alloc((size_t)MLPD * D * 2);
  bf16_t* woutb= (bf16_t*)alloc((size_t)D * MLPD * 2);

  const int nqkv = H * DH * DH;   // 49152
  const int nmlp = MLPD * D;      // 2359296
  cvt_f32_bf16<<<(nqkv + 255) / 256, 256, 0, stream>>>(Wq, wqb, nqkv);
  cvt_f32_bf16<<<(nqkv + 255) / 256, 256, 0, stream>>>(Wk, wkb, nqkv);
  cvt_f32_bf16<<<(nqkv + 255) / 256, 256, 0, stream>>>(Wv, wvb, nqkv);
  cvt_f32_bf16<<<(nmlp + 255) / 256, 256, 0, stream>>>(W_in, winb, nmlp);
  cvt_f32_bf16<<<(nmlp + 255) / 256, 256, 0, stream>>>(W_out, woutb, nmlp);

  resid_ln_kernel<<<ROWS, 256, 0, stream>>>(patch, nullptr, nullptr, g1, b1, xln);
  qkv_kernel<<<dim3(ROWS / 64, H), 128, 0, stream>>>(xln, wqb, wkb, wvb, bq, bk, bv, qb, kb, vb);
  attn_kernel<<<dim3(Bsz * H, S / 128), 256, 0, stream>>>(qb, kb, vb, attn);
  resid_ln_kernel<<<ROWS, 256, 0, stream>>>(patch, attn, out, g2, b2, h2);
  mlp1_kernel<<<dim3(ROWS / 128, MLPD / 128), 256, 0, stream>>>(h2, winb, b_in, hmid);
  mlp2_kernel<<<dim3(ROWS / 128, D / 64), 256, 0, stream>>>(hmid, woutb, b_out, out);
}
